// ELMoEncoder_83691732730258
// MI455X (gfx1250) — compile-verified, matrix operand order checked by reading, not run on hardware
//
#include <hip/hip_runtime.h>
#include <hip/hip_bf16.h>

// ---------------------------------------------------------------------------
// ELMo encoder forward for gfx1250 (MI455X).
// All heavy GEMMs + LSTM recurrences use v_wmma_f32_16x16x32_bf16 (bf16 in,
// f32 accumulate). Weights are pre-converted once per launch to transposed
// bf16 [N][K] so the GEMM inner loops do zero conversion work.
// ---------------------------------------------------------------------------

typedef __attribute__((ext_vector_type(16))) __bf16 bf16x16;
typedef __attribute__((ext_vector_type(8)))  __bf16 bf16x8;
typedef __attribute__((ext_vector_type(8)))  float  f32x8;
typedef __attribute__((ext_vector_type(4)))  float  f32x4;

#define DEV __device__ __forceinline__

DEV float sigmoidf_(float x) { return 1.0f / (1.0f + __expf(-x)); }
DEV float gelu_(float x) {
    float x3 = x * x * x;
    return 0.5f * x * (1.0f + tanhf(0.7978845608028654f * (x + 0.044715f * x3)));
}

// Build a v16bf WMMA operand from two runs of 8 contiguous bf16 values
// (per-lane layout: slots 0..7 -> k = base+0..7, slots 8..15 -> k = base+16..23).
DEV bf16x16 pack16(const __bf16* lo, const __bf16* hi) {
    bf16x8 a = *(const bf16x8*)lo;
    bf16x8 b = *(const bf16x8*)hi;
    bf16x16 r;
#pragma unroll
    for (int j = 0; j < 8; ++j) { r[j] = a[j]; r[j + 8] = b[j]; }
    return r;
}

// Convert 8 contiguous f32 to a bf16x8 (lowers to packed cvt pairs).
DEV bf16x8 cvt8(const float* p) {
    f32x4 a = *(const f32x4*)p;
    f32x4 b = *(const f32x4*)(p + 4);
    bf16x8 r;
#pragma unroll
    for (int j = 0; j < 4; ++j) { r[j] = (__bf16)a[j]; r[j + 4] = (__bf16)b[j]; }
    return r;
}

// ---------------------------------------------------------------------------
// bf16-WMMA GEMM:  C[M,N] = A[M,K] * Bt^T + bias
//   A  : f32 [M,K] row-major (row stride lda)
//   Bt : bf16 [N,K] row-major (pre-transposed weights)
// Block: 256 threads (8 waves), tile 64(M) x 128(N), K-tile 64; wave tile 32x32.
// Requires M%64==0, N%128==0, K%64==0 (true for every GEMM here).
// ---------------------------------------------------------------------------
__global__ __launch_bounds__(256) void gemm_bf16_wmma(
    const float* __restrict__ A, int lda,
    const __bf16* __restrict__ Bt,
    const float* __restrict__ bias,
    float* __restrict__ C, int ldc, int K)
{
    __shared__ __bf16 As[64 * 64];    //  8 KB, [m][k]
    __shared__ __bf16 Bs[128 * 64];   // 16 KB, [n][k]

    const int tid  = threadIdx.x;
    const int wave = tid >> 5, lane = tid & 31;
    const int wm = wave >> 2, wn = wave & 3;          // 2 x 4 wave grid
    const int m0 = blockIdx.y * 64, n0 = blockIdx.x * 128;
    const int lm = lane & 15, half = lane >> 4;

    f32x8 acc[2][2] = {};

    for (int k0 = 0; k0 < K; k0 += 64) {
        // ---- stage A: 64x64 f32 -> bf16, 16 elems / thread, vector ops ----
#pragma unroll
        for (int i = 0; i < 2; ++i) {
            int c8 = tid + i * 256;                    // 512 chunks of 8
            int r = c8 >> 3, c = (c8 & 7) * 8;
            const float* src = A + (size_t)(m0 + r) * lda + k0 + c;
            *(bf16x8*)&As[r * 64 + c] = cvt8(src);
            if (k0 + 64 < K) __builtin_prefetch(src + 64, 0, 1);
        }
        // ---- stage B: 128x64 bf16, raw 16B copies, 32 elems / thread ----
#pragma unroll
        for (int i = 0; i < 4; ++i) {
            int c8 = tid + i * 256;                    // 1024 chunks of 8
            int nn = c8 >> 3, c = (c8 & 7) * 8;
            const __bf16* src = Bt + (size_t)(n0 + nn) * K + k0 + c;
            *(bf16x8*)&Bs[nn * 64 + c] = *(const bf16x8*)src;
            if (k0 + 64 < K) __builtin_prefetch(src + 64, 0, 1);
        }
        __syncthreads();
        // ---- compute: 8 WMMA per barrier pair ----
#pragma unroll
        for (int kc = 0; kc < 64; kc += 32) {
#pragma unroll
            for (int mt = 0; mt < 2; ++mt) {
                const __bf16* arow = &As[(wm * 32 + mt * 16 + lm) * 64 + kc + half * 8];
                bf16x16 av = pack16(arow, arow + 16);
#pragma unroll
                for (int nt = 0; nt < 2; ++nt) {
                    const __bf16* brow = &Bs[(wn * 32 + nt * 16 + lm) * 64 + kc + half * 8];
                    bf16x16 bv = pack16(brow, brow + 16);
                    acc[mt][nt] = __builtin_amdgcn_wmma_f32_16x16x32_bf16(
                        false, av, false, bv, (short)0, acc[mt][nt], false, false);
                }
            }
        }
        __syncthreads();
    }

#pragma unroll
    for (int mt = 0; mt < 2; ++mt)
#pragma unroll
        for (int nt = 0; nt < 2; ++nt) {
            int col = n0 + wn * 32 + nt * 16 + lm;
            float bv = bias ? bias[col] : 0.0f;
#pragma unroll
            for (int v = 0; v < 8; ++v) {
                int row = m0 + wm * 32 + mt * 16 + v + half * 8;
                C[(size_t)row * ldc + col] = acc[mt][nt][v] + bv;
            }
        }
}

// ---------------------------------------------------------------------------
// LSTM layer (Keras gate order i,f,g,o). One block per chain (grid=2),
// 1024 threads = 32 waves. Wave w owns gate columns [16w,16w+16) of all four
// gates, so the cell state lives in registers (accumulator layout). h kept in
// LDS as bf16 (WMMA A operand); U pre-transposed to bf16 [4H][H].
// xw layout: [B*T, 4H] row-major, row = b*256 + t.
// ---------------------------------------------------------------------------
__global__ __launch_bounds__(1024) void lstm_layer_kernel(
    const float* __restrict__ xw_f, const float* __restrict__ xw_b,
    const __bf16* __restrict__ Ut_f, const __bf16* __restrict__ Ut_b,
    float* __restrict__ out_f, float* __restrict__ out_b,
    int ostr, int ocol_f, int ocol_b)
{
    __shared__ __bf16 hbf[16 * 512];                   // h as bf16, [B=16][H=512]

    const int chain = blockIdx.x;
    const float*  xw  = chain ? xw_b  : xw_f;
    const __bf16* Ut  = chain ? Ut_b  : Ut_f;
    float*        out = chain ? out_b : out_f;
    const int     ocol = chain ? ocol_b : ocol_f;

    const int tid  = threadIdx.x;
    const int wave = tid >> 5, lane = tid & 31;
    const int lm = lane & 15, half = lane >> 4;
    const int j = wave * 16 + lm;                      // column within H

    for (int i = tid; i < 16 * 512; i += 1024) hbf[i] = (__bf16)0.0f;
    __syncthreads();

    f32x8 cst = {};                                    // cell state (registers)

    for (int t = 0; t < 256; ++t) {
        f32x8 acc[4];
#pragma unroll
        for (int g = 0; g < 4; ++g) {                  // seed with x@W + b
            int col = g * 512 + j;
#pragma unroll
            for (int v = 0; v < 8; ++v) {
                int m = v + half * 8;
                acc[g][v] = xw[(size_t)(m * 256 + t) * 2048 + col];
            }
            if (t + 1 < 256)                           // pull next step into cache
                __builtin_prefetch(&xw[(size_t)(half * 8 * 256 + t + 1) * 2048 + col], 0, 1);
        }
        for (int k0 = 0; k0 < 512; k0 += 32) {         // += h @ U
            const __bf16* ap = &hbf[lm * 512 + k0 + half * 8];
            bf16x16 av = pack16(ap, ap + 16);
#pragma unroll
            for (int g = 0; g < 4; ++g) {
                int n = g * 512 + j;
                const __bf16* bp = &Ut[(size_t)n * 512 + k0 + half * 8];
                bf16x16 bv = pack16(bp, bp + 16);
                acc[g] = __builtin_amdgcn_wmma_f32_16x16x32_bf16(
                    false, av, false, bv, (short)0, acc[g], false, false);
            }
        }
        f32x8 hv;
#pragma unroll
        for (int v = 0; v < 8; ++v) {
            float iv = sigmoidf_(acc[0][v]);
            float fv = sigmoidf_(acc[1][v]);
            float gv = tanhf(acc[2][v]);
            float ov = sigmoidf_(acc[3][v]);
            float cc = fv * cst[v] + iv * gv;
            cst[v] = cc;
            hv[v] = ov * tanhf(cc);
        }
        __syncthreads();                               // all waves done reading h
#pragma unroll
        for (int v = 0; v < 8; ++v) {
            int m = v + half * 8;
            hbf[m * 512 + j] = (__bf16)hv[v];
            out[(size_t)(m * 256 + t) * ostr + ocol + j] = hv[v];
        }
        __syncthreads();
    }
}

// ---------------------------------------------------------------------------
// W [K][N] f32 (row-major)  ->  Wt [N][K] bf16   (one-time weight conversion)
// ---------------------------------------------------------------------------
__global__ __launch_bounds__(256) void transpose_to_bf16(
    const float* __restrict__ W, __bf16* __restrict__ Wt, int K, int N)
{
    int idx = blockIdx.x * 256 + threadIdx.x;
    if (idx >= K * N) return;
    int k = idx / N, n = idx % N;
    Wt[(size_t)n * K + k] = (__bf16)W[idx];
}

// ---------------------------------------------------------------------------
// CharCNN: one block per word. ce[16][64] in LDS; each thread owns 8 filters;
// conv weights read once per filter (positions innermost); gelu + max-pool.
// ---------------------------------------------------------------------------
struct ConvArgs { const float* k[7]; const float* b[7]; };

__global__ __launch_bounds__(256) void charcnn_kernel(
    const int* __restrict__ char_ids, const float* __restrict__ char_emb,
    ConvArgs cv, float* __restrict__ Wout)
{
    __shared__ float ce[16 * 64];
    __shared__ int   ids[16];
    const int word = blockIdx.x;
    const int tid  = threadIdx.x;

    if (tid < 16) ids[tid] = char_ids[word * 16 + tid];
    __syncthreads();
    for (int i = tid; i < 1024; i += 256) {
        int c = i >> 6, d = i & 63;
        ce[i] = char_emb[(size_t)ids[c] * 64 + d];
    }
    __syncthreads();

    for (int gf = tid; gf < 2048; gf += 256) {
        int wi, fl, fc;
        if      (gf < 32)  { wi = 0; fl = gf;        fc = 32;   }
        else if (gf < 64)  { wi = 1; fl = gf - 32;   fc = 32;   }
        else if (gf < 128) { wi = 2; fl = gf - 64;   fc = 64;   }
        else if (gf < 256) { wi = 3; fl = gf - 128;  fc = 128;  }
        else if (gf < 512) { wi = 4; fl = gf - 256;  fc = 256;  }
        else if (gf < 1024){ wi = 5; fl = gf - 512;  fc = 512;  }
        else               { wi = 6; fl = gf - 1024; fc = 1024; }
        const int wdt = wi + 1;
        const int np  = 16 - wdt + 1;
        const float* kk = cv.k[wi];
        const float  bb = cv.b[wi][fl];

        float accp[16];
#pragma unroll
        for (int p = 0; p < 16; ++p) accp[p] = bb;
        for (int i = 0; i < wdt; ++i)
            for (int d = 0; d < 64; ++d) {
                float wv = kk[(size_t)(i * 64 + d) * fc + fl];
                const float* cep = &ce[i * 64 + d];
#pragma unroll
                for (int p = 0; p < 16; ++p)
                    if (p < np) accp[p] += cep[p * 64] * wv;
            }
        float mx = -3.4e38f;
#pragma unroll
        for (int p = 0; p < 16; ++p)
            if (p < np) mx = fmaxf(mx, gelu_(accp[p]));
        Wout[(size_t)word * 2048 + gf] = mx;
    }
}

// w = sigmoid(tpre)*gelu(hpre) + (1-sigmoid(tpre))*w
__global__ void highway_combine(float* __restrict__ w,
                                const float* __restrict__ tpre,
                                const float* __restrict__ hpre, int n)
{
    int idx = blockIdx.x * blockDim.x + threadIdx.x;
    if (idx < n) {
        float t = sigmoidf_(tpre[idx]);
        float g = gelu_(hpre[idx]);
        float x = w[idx];
        w[idx] = t * g + (1.0f - t) * x;
    }
}

// In-place LayerNorm over rows of 1024, eps = 1e-12
__global__ __launch_bounds__(256) void layernorm_kernel(
    float* __restrict__ e, const float* __restrict__ gamma,
    const float* __restrict__ beta)
{
    __shared__ float red[256];
    __shared__ float red2[256];
    const int row = blockIdx.x, tid = threadIdx.x;
    float* x = e + (size_t)row * 1024;
    float s = 0.f, s2 = 0.f, vals[4];
#pragma unroll
    for (int i = 0; i < 4; ++i) {
        float v = x[tid + i * 256];
        vals[i] = v; s += v; s2 += v * v;
    }
    red[tid] = s; red2[tid] = s2;
    __syncthreads();
    for (int off = 128; off > 0; off >>= 1) {
        if (tid < off) { red[tid] += red[tid + off]; red2[tid] += red2[tid + off]; }
        __syncthreads();
    }
    float mu  = red[0] * (1.0f / 1024.0f);
    float var = red2[0] * (1.0f / 1024.0f) - mu * mu;
    float rstd = rsqrtf(var + 1e-12f);
#pragma unroll
    for (int i = 0; i < 4; ++i) {
        int c = tid + i * 256;
        x[c] = gamma[c] * (vals[i] - mu) * rstd + beta[c];
    }
}

// pooled[b][j] = tanh(seq[b,0,:] . pool_W[:,j] + pool_b[j]);  8192 threads
__global__ __launch_bounds__(256) void pool_kernel(
    const float* __restrict__ seq, const float* __restrict__ pw,
    const float* __restrict__ pb, float* __restrict__ pooled)
{
    int idx = blockIdx.x * 256 + threadIdx.x;
    int b = idx >> 9, jj = idx & 511;
    const float* xr = seq + (size_t)b * 256 * 1024;
    float acc = pb[jj];
    for (int k = 0; k < 1024; ++k) acc += xr[k] * pw[(size_t)k * 512 + jj];
    pooled[idx] = tanhf(acc);
}

// ---------------------------------------------------------------------------
extern "C" void kernel_launch(void* const* d_in, const int* in_sizes, int n_in,
                              void* d_out, int out_size, void* d_ws, size_t ws_size,
                              hipStream_t stream)
{
    (void)in_sizes; (void)n_in; (void)out_size; (void)ws_size;

    // Inputs: depth-first flatten of setup_inputs() dict (insertion order)
    const int*   char_ids = (const int*)d_in[0];
    /* d_in[1] = mask (unused by the math) */
    const float* char_emb = (const float*)d_in[2];
    ConvArgs cv;
    for (int i = 0; i < 7; ++i) {
        cv.k[i] = (const float*)d_in[3 + 2 * i];
        cv.b[i] = (const float*)d_in[4 + 2 * i];
    }
    const float* hw_Wh[2] = {(const float*)d_in[17], (const float*)d_in[21]};
    const float* hw_bh[2] = {(const float*)d_in[18], (const float*)d_in[22]};
    const float* hw_Wt[2] = {(const float*)d_in[19], (const float*)d_in[23]};
    const float* hw_bt[2] = {(const float*)d_in[20], (const float*)d_in[24]};
    const float* proj_W = (const float*)d_in[25];
    const float* proj_b = (const float*)d_in[26];
    const float* ln_g   = (const float*)d_in[27];
    const float* ln_b   = (const float*)d_in[28];
    const float *lstm_W[4], *lstm_U[4], *lstm_b[4];
    for (int i = 0; i < 4; ++i) {           // [fwd_l0, fwd_l1, bwd_l0, bwd_l1]
        lstm_W[i] = (const float*)d_in[29 + 3 * i];
        lstm_U[i] = (const float*)d_in[30 + 3 * i];
        lstm_b[i] = (const float*)d_in[31 + 3 * i];
    }
    const float* pool_W = (const float*)d_in[41];
    const float* pool_b = (const float*)d_in[42];
    const float* out_W  = (const float*)d_in[43];
    const float* out_b  = (const float*)d_in[44];

    // Outputs (concatenated): seq | pooled | fwd_logits | bwd_logits
    float* out     = (float*)d_out;
    float* seq     = out;                                  // 16*256*1024
    float* pooled  = out + 4194304;                        // 16*512
    float* fwd_log = out + 4194304 + 8192;                 // 16*256*16384
    float* bwd_log = fwd_log + 67108864;

    // Workspace layout
    float* ws   = (float*)d_ws;
    float* wbuf = ws;                   // [4096,2048] charcnn/highway activations
    float* tmp1 = ws + 8388608;         // [4096,2048] t-gate pre / xw fwd
    float* tmp2 = ws + 16777216;        // [4096,2048] h-gate pre / xw bwd
    float* ebuf = ws + 25165824;        // [4096,1024] projection -> LN (in place)
    float* h0f  = ws + 29360128;        // [4096,512] layer0 fwd h
    float* h0b  = ws + 31457280;        // [4096,512] layer0 bwd h

    // bf16 transposed-weight region
    __bf16* bw = (__bf16*)(ws + 33554432);
    __bf16* UtT[4];                     // [2048][512]
    for (int i = 0; i < 4; ++i) UtT[i] = bw + (size_t)i * 1048576;
    __bf16* hwWtT[2] = { bw +  4194304, bw + 12582912 };   // [2048][2048]
    __bf16* hwWhT[2] = { bw +  8388608, bw + 16777216 };   // [2048][2048]
    __bf16* projT    =   bw + 20971520;                    // [1024][2048]
    __bf16* lstmWT[4];
    lstmWT[0] = bw + 23068672;                             // [2048][1024]
    lstmWT[1] = bw + 25165824;                             // [2048][512]
    lstmWT[2] = bw + 26214400;                             // [2048][1024]
    lstmWT[3] = bw + 28311552;                             // [2048][512]
    __bf16* outWT    =   bw + 29360128;                    // [16384][512]

    dim3 b256(256);

    // 0. one-time weight conversions -> transposed bf16 [N][K]
    for (int i = 0; i < 4; ++i)
        transpose_to_bf16<<<4096, b256, 0, stream>>>(lstm_U[i], UtT[i], 512, 2048);
    for (int l = 0; l < 2; ++l) {
        transpose_to_bf16<<<16384, b256, 0, stream>>>(hw_Wt[l], hwWtT[l], 2048, 2048);
        transpose_to_bf16<<<16384, b256, 0, stream>>>(hw_Wh[l], hwWhT[l], 2048, 2048);
    }
    transpose_to_bf16<<<8192, b256, 0, stream>>>(proj_W, projT, 2048, 1024);
    transpose_to_bf16<<<8192, b256, 0, stream>>>(lstm_W[0], lstmWT[0], 1024, 2048);
    transpose_to_bf16<<<4096, b256, 0, stream>>>(lstm_W[1], lstmWT[1], 512, 2048);
    transpose_to_bf16<<<8192, b256, 0, stream>>>(lstm_W[2], lstmWT[2], 1024, 2048);
    transpose_to_bf16<<<4096, b256, 0, stream>>>(lstm_W[3], lstmWT[3], 512, 2048);
    transpose_to_bf16<<<32768, b256, 0, stream>>>(out_W, outWT, 512, 16384);

    // 1. char CNN -> wbuf [4096,2048]
    charcnn_kernel<<<4096, b256, 0, stream>>>(char_ids, char_emb, cv, wbuf);

    // 2. highway x2
    for (int l = 0; l < 2; ++l) {
        gemm_bf16_wmma<<<dim3(16, 64), b256, 0, stream>>>(wbuf, 2048, hwWtT[l], hw_bt[l], tmp1, 2048, 2048);
        gemm_bf16_wmma<<<dim3(16, 64), b256, 0, stream>>>(wbuf, 2048, hwWhT[l], hw_bh[l], tmp2, 2048, 2048);
        highway_combine<<<32768, b256, 0, stream>>>(wbuf, tmp1, tmp2, 8388608);
    }

    // 3. projection to 2H + LayerNorm (in place)
    gemm_bf16_wmma<<<dim3(8, 64), b256, 0, stream>>>(wbuf, 2048, projT, proj_b, ebuf, 1024, 2048);
    layernorm_kernel<<<4096, b256, 0, stream>>>(ebuf, ln_g, ln_b);

    // 4. layer0 input projections (both chains), then layer0 LSTMs
    gemm_bf16_wmma<<<dim3(16, 64), b256, 0, stream>>>(ebuf, 1024, lstmWT[0], lstm_b[0], tmp1, 2048, 1024);
    gemm_bf16_wmma<<<dim3(16, 64), b256, 0, stream>>>(ebuf, 1024, lstmWT[2], lstm_b[2], tmp2, 2048, 1024);
    lstm_layer_kernel<<<2, 1024, 0, stream>>>(tmp1, tmp2, UtT[0], UtT[2], h0f, h0b, 512, 0, 0);

    // 5. layer1 input projections, then layer1 LSTMs writing into seq halves
    gemm_bf16_wmma<<<dim3(16, 64), b256, 0, stream>>>(h0f, 512, lstmWT[1], lstm_b[1], tmp1, 2048, 512);
    gemm_bf16_wmma<<<dim3(16, 64), b256, 0, stream>>>(h0b, 512, lstmWT[3], lstm_b[3], tmp2, 2048, 512);
    lstm_layer_kernel<<<2, 1024, 0, stream>>>(tmp1, tmp2, UtT[1], UtT[3], seq, seq, 1024, 0, 512);

    // 6. pooled = tanh(seq[:,0,:] @ pool_W + pool_b)
    pool_kernel<<<32, b256, 0, stream>>>(seq, pool_W, pool_b, pooled);

    // 7. logits: f_half / b_half @ out_W + out_b
    gemm_bf16_wmma<<<dim3(128, 64), b256, 0, stream>>>(seq,       1024, outWT, out_b, fwd_log, 16384, 512);
    gemm_bf16_wmma<<<dim3(128, 64), b256, 0, stream>>>(seq + 512, 1024, outWT, out_b, bwd_log, 16384, 512);
}
